// MyMultilevelDense_32512902431060
// MI455X (gfx1250) — matrix operand order repaired and checked by reading, not compiled, not run on hardware
//
#include <hip/hip_runtime.h>
#include <hip/hip_bf16.h>
#include <math.h>

// ---------------- problem constants ----------------
#define BATCH     2048
#define UNITS     256
#define FEATURES  256
#define NGROUPS   64
#define SOFTC     0.5413248546129181f   // log(expm1(1.0))

typedef __attribute__((ext_vector_type(16))) _Float16 v16h;
typedef __attribute__((ext_vector_type(8)))  float    v8f;
typedef __attribute__((ext_vector_type(4)))  unsigned int u32x4;
typedef __attribute__((ext_vector_type(4)))  int      i32x4;
typedef __attribute__((ext_vector_type(8)))  int      i32x8;

#if defined(__has_builtin)
#if __has_builtin(__builtin_amdgcn_tensor_load_to_lds) && \
    __has_builtin(__builtin_amdgcn_s_wait_tensorcnt)
#define HAVE_TDM 1
#endif
#endif
#ifndef HAVE_TDM
#define HAVE_TDM 0
#endif

__device__ __forceinline__ float softplus_f(float v) {
    return (v > 20.0f) ? v : log1pf(__expf(v));
}

// ---------------- workspace layout (ints) ----------------
#define WS_COUNTS  0
#define WS_CURSORS 64
#define WS_OFFSETS 128
#define WS_ROWIDS  200

// ---------------- binning kernels ----------------
__global__ void mk_zero(int* ws) {
    int t = threadIdx.x;
    if (t < 128) ws[t] = 0;   // counts + cursors
}

__global__ void mk_count(const int* __restrict__ gid, int* ws) {
    int b = blockIdx.x * blockDim.x + threadIdx.x;
    if (b < BATCH) atomicAdd(&ws[WS_COUNTS + gid[b]], 1);
}

__global__ void mk_scan(int* ws) {
    if (threadIdx.x == 0) {
        int acc = 0;
        for (int g = 0; g < NGROUPS; ++g) {
            ws[WS_OFFSETS + g] = acc;
            acc += ws[WS_COUNTS + g];
        }
        ws[WS_OFFSETS + NGROUPS] = acc;
    }
}

__global__ void mk_scatter(const int* __restrict__ gid, int* ws) {
    int b = blockIdx.x * blockDim.x + threadIdx.x;
    if (b < BATCH) {
        int g = gid[b];
        int pos = ws[WS_OFFSETS + g] + atomicAdd(&ws[WS_CURSORS + g], 1);
        ws[WS_ROWIDS + pos] = b;
    }
}

// ---------------- grouped WMMA GEMM: out[b,u] = sum_p w_mu[g,u,p]*x[b,p] ----
// grid = (NGROUPS, 4), block = 128 (4 waves; wave w handles u-tile blockIdx.y*4+w)
// x-row gather per 16-row chunk staged to LDS via TDM gather-mode descriptor.
__global__ void __launch_bounds__(128)
mk_group_gemm(const float* __restrict__ x,
              const float* __restrict__ w_mu,
              const int*   __restrict__ ws,
              float*       __restrict__ out) {
    __shared__ float xs[16 * FEATURES];   // 16 gathered rows of x (16 KB)
    __shared__ int   rids[16];

    const int g     = blockIdx.x;
    const int tid   = threadIdx.x;
    const int wave  = tid >> 5;
    const int lane  = tid & 31;
    const int m     = lane & 15;          // A-row / B-col / D-col index
    const int hi    = lane >> 4;          // lane-half selector
    const int u0    = (blockIdx.y * 4 + wave) * 16;

    const int start = ws[WS_OFFSETS + g];
    const int nrows = ws[WS_OFFSETS + g + 1] - start;
    const int nchunks = (nrows + 15) >> 4;

    for (int ch = 0; ch < nchunks; ++ch) {
        // ---- stage row ids ----
        if (tid < 16) {
            int r = ch * 16 + tid;
            rids[tid] = (r < nrows) ? ws[WS_ROWIDS + start + r] : -1;
        }
        __syncthreads();

#if HAVE_TDM
        // ---- TDM gather: 16 rows x 256 f32 from x -> LDS, one instruction ----
        if (wave == 0) {
            unsigned int idx[16];
            #pragma unroll
            for (int j = 0; j < 16; ++j) {
                int r = ch * 16 + j;
                int rid = (r < nrows) ? ws[WS_ROWIDS + start + r] : 0;
                idx[j] = (unsigned int)__builtin_amdgcn_readfirstlane(rid) & 0xFFFFu;
            }
            unsigned long long ga = (unsigned long long)(uintptr_t)x;
            unsigned int lds_off  = (unsigned int)(uintptr_t)(void*)xs;
            // D# group 0: count=1, gather_mode=1 (16-bit idx), lds addr,
            //             57-bit global addr, type=2 ("image")
            u32x4 g0 = { 1u | (1u << 31),
                         lds_off,
                         (unsigned int)(ga & 0xFFFFFFFFull),
                         (unsigned int)((ga >> 32) & 0x1FFFFFFull) | (2u << 30) };
            // D# group 1: data_size=4B; tensor_dim0=256; tensor_dim1=BATCH;
            //             tile_dim0=256; tile_dim1=16 indices; dim0_stride=256
            i32x8 g1 = { (int)(2u << 16),
                         (int)(256u << 16),
                         (int)((unsigned)BATCH << 16),
                         (int)(256u << 16),
                         16,
                         256,
                         0, 0 };
            // D# groups 2/3: sixteen 16-bit row indices
            i32x4 g2 = { (int)(idx[0]  | (idx[1]  << 16)),
                         (int)(idx[2]  | (idx[3]  << 16)),
                         (int)(idx[4]  | (idx[5]  << 16)),
                         (int)(idx[6]  | (idx[7]  << 16)) };
            i32x4 g3 = { (int)(idx[8]  | (idx[9]  << 16)),
                         (int)(idx[10] | (idx[11] << 16)),
                         (int)(idx[12] | (idx[13] << 16)),
                         (int)(idx[14] | (idx[15] << 16)) };
#if defined(__clang_major__) && (__clang_major__ >= 23)
            i32x8 gz = { 0, 0, 0, 0, 0, 0, 0, 0 };
            __builtin_amdgcn_tensor_load_to_lds(g0, g1, g2, g3, gz, 0);
#else
            __builtin_amdgcn_tensor_load_to_lds(g0, g1, g2, g3, 0);
#endif
            __builtin_amdgcn_s_wait_tensorcnt(0);
        }
        __syncthreads();
#else
        // ---- fallback: cooperative float4 staging ----
        #pragma unroll
        for (int i = 0; i < 8; ++i) {
            int idx  = tid + i * 128;
            int row  = idx >> 6;
            int col4 = idx & 63;
            int rid  = rids[row];
            int src  = (rid >= 0) ? rid : 0;
            ((float4*)xs)[row * 64 + col4] =
                ((const float4*)x)[(size_t)src * 64 + col4];
        }
        __syncthreads();
#endif

        // ---- WMMA over K=FEATURES in steps of 32 (split-f16 -> ~f32) ----
        v8f c = {};
        for (int p0 = 0; p0 < FEATURES; p0 += 32) {
            v16h a_hi, a_lo, b_hi, b_lo;
            #pragma unroll
            for (int i = 0; i < 8; ++i) {
                float xv0 = xs[m * FEATURES + p0 + hi * 8 + i];
                _Float16 h0 = (_Float16)xv0;
                a_hi[i] = h0;
                a_lo[i] = (_Float16)(xv0 - (float)h0);
                float xv1 = xs[m * FEATURES + p0 + 16 + hi * 8 + i];
                _Float16 h1 = (_Float16)xv1;
                a_hi[8 + i] = h1;
                a_lo[8 + i] = (_Float16)(xv1 - (float)h1);
            }
            const float* wrow =
                w_mu + ((size_t)g * UNITS + (u0 + m)) * FEATURES + p0 + hi * 8;
            #pragma unroll
            for (int i = 0; i < 16; ++i) {
                float wv = wrow[(i & 7) + ((i >> 3) << 4)];
                _Float16 h = (_Float16)wv;
                b_hi[i] = h;
                b_lo[i] = (_Float16)(wv - (float)h);
            }
            c = __builtin_amdgcn_wmma_f32_16x16x32_f16(false, a_hi, false, b_hi,
                                                       (short)0, c, false, false);
            c = __builtin_amdgcn_wmma_f32_16x16x32_f16(false, a_lo, false, b_hi,
                                                       (short)0, c, false, false);
            c = __builtin_amdgcn_wmma_f32_16x16x32_f16(false, a_hi, false, b_lo,
                                                       (short)0, c, false, false);
        }

        // ---- store D: VGPR r, lane -> (M=r+8*hi, N=m) ----
        #pragma unroll
        for (int r = 0; r < 8; ++r) {
            int mm  = r + 8 * hi;
            int rid = rids[mm];
            if (rid >= 0) out[(size_t)rid * UNITS + u0 + m] = c[r];
        }
        __syncthreads();
    }
}

// ---------------- eps streaming epilogue (HBM-bound, sets runtime) ----------
__global__ void __launch_bounds__(256)
mk_eps_stream(const float* __restrict__ x,
              const int*   __restrict__ gid,
              const float* __restrict__ w_sigma,
              const float* __restrict__ b_mu,
              const float* __restrict__ b_sigma,
              const float* __restrict__ eps_w,
              const float* __restrict__ eps_b,
              float*       __restrict__ out) {
    __shared__ float xl[FEATURES];
    const int b    = blockIdx.x;
    const int tid  = threadIdx.x;
    const int wave = tid >> 5;
    const int lane = tid & 31;
    const int u    = blockIdx.y * 8 + wave;
    const int g    = gid[b];

    xl[tid] = x[(size_t)b * FEATURES + tid];
    __syncthreads();

    const size_t erow = ((size_t)b * UNITS + u) * FEATURES;
    const size_t srow = ((size_t)g * UNITS + u) * FEATURES;

    float acc = 0.0f;
    #pragma unroll
    for (int it = 0; it < 2; ++it) {
        int p = it * 128 + lane * 4;
        float4 e  = ((const float4*)(eps_w   + erow))[p >> 2];
        float4 s  = ((const float4*)(w_sigma + srow))[p >> 2];
        float4 xv = ((const float4*)xl)[p >> 2];
        acc += softplus_f(SOFTC + s.x) * e.x * xv.x;
        acc += softplus_f(SOFTC + s.y) * e.y * xv.y;
        acc += softplus_f(SOFTC + s.z) * e.z * xv.z;
        acc += softplus_f(SOFTC + s.w) * e.w * xv.w;
    }
    #pragma unroll
    for (int off = 16; off > 0; off >>= 1)
        acc += __shfl_down(acc, off, 32);

    if (lane == 0) {
        float bm = b_mu[g * UNITS + u];
        float bs = b_sigma[g * UNITS + u];
        float eb = eps_b[(size_t)b * UNITS + u];
        out[(size_t)b * UNITS + u] += acc + bm + softplus_f(SOFTC + bs) * eb;
    }
}

// ---------------- launcher ----------------
extern "C" void kernel_launch(void* const* d_in, const int* in_sizes, int n_in,
                              void* d_out, int out_size, void* d_ws, size_t ws_size,
                              hipStream_t stream) {
    const float* x       = (const float*)d_in[0];
    const int*   gid     = (const int*)  d_in[1];
    const float* w_mu    = (const float*)d_in[2];
    const float* w_sigma = (const float*)d_in[3];
    const float* b_mu    = (const float*)d_in[4];
    const float* b_sigma = (const float*)d_in[5];
    const float* eps_w   = (const float*)d_in[6];
    const float* eps_b   = (const float*)d_in[7];
    float* out = (float*)d_out;
    int*   ws  = (int*)d_ws;

    mk_zero   <<<1, 128, 0, stream>>>(ws);
    mk_count  <<<BATCH / 256, 256, 0, stream>>>(gid, ws);
    mk_scan   <<<1, 32, 0, stream>>>(ws);
    mk_scatter<<<BATCH / 256, 256, 0, stream>>>(gid, ws);

    mk_group_gemm<<<dim3(NGROUPS, 4), 128, 0, stream>>>(x, w_mu, ws, out);

    mk_eps_stream<<<dim3(BATCH, UNITS / 8), 256, 0, stream>>>(
        x, gid, w_sigma, b_mu, b_sigma, eps_w, eps_b, out);
}
